// CellLatentPerturbation_39779987096432
// MI455X (gfx1250) — matrix operand order repaired and checked by reading, not compiled, not run on hardware
//
#include <hip/hip_runtime.h>
#include <cstdint>
#include <cstddef>

typedef __attribute__((ext_vector_type(16))) _Float16 v16h;
typedef __attribute__((ext_vector_type(8)))  float    v8f;

// ---------------------------------------------------------------------------
// WMMA GEMM: C[M,Nc] = A[M,K] @ B[K,Nc] (+bias). B is pre-transposed,
// pre-converted f16: Bt[Nc][K]. A is f16 [M][K] (A16=true) or f32 (A16=false).
// f32 accumulate via v_wmma_f32_16x16x32_f16.
// Block = 256 threads (8 waves). Tile BM=128 x BN=64, K-step 32.
// Wave w: 32x32 macro-tile at rows (w&3)*32, cols (w>>2)*32 -> 4 WMMA/step.
// A16 path (K==256): persistent 32KB B panel in LDS (loaded once, async) +
// double-buffered A tiles streamed with ASYNCcnt overlap.
// f32 path: per-step VALU convert of A + double-buffered async B tiles.
// EPI 0: C = AB + bias ; 1: C = relu(AB + bias) ; 2: C += AB + bias ; 3: C = AB
// ---------------------------------------------------------------------------
template <int EPI, bool A16>
__global__ __launch_bounds__(256) void wmma_gemm(
    const void* __restrict__ Av, const _Float16* __restrict__ Bt,
    const float* __restrict__ bias, float* __restrict__ C,
    int M, int K, int Nc) {
  __shared__ __align__(16) _Float16 sA[2 * 128 * 32]; // double-buffered A tiles
  __shared__ __align__(16) _Float16 sB[64 * 256];     // A16: full B panel [n][k]
                                                      // f32: two 64x32 buffers
  const int m0 = blockIdx.x * 128;
  const int n0 = blockIdx.y * 64;
  const int tid = threadIdx.x;
  const int wid = tid >> 5, lane = tid & 31;
  const int wm = (wid & 3) * 32;
  const int wn = (wid >> 2) * 32;
  v8f c00 = {}, c01 = {}, c10 = {}, c11 = {};

  // Fragment addressing per ISA layouts (wave32):
  // A 16x32 f16: lanes 0-15 -> M=lane, halves K=[0..7],[16..23]; lanes 16-31 -> K=[8..15],[24..31]
  // B 32x16 f16: lanes 0-15 -> N=lane, halves K=[0..7],[16..23]; lanes 16-31 -> K=[8..15],[24..31]
  const int lh  = lane & 15;
  const int kbA = (lane < 16) ? 0 : 8;
  const int kbB = (lane < 16) ? 0 : 8;
  const int mA0 = wm + lh;
  const int mA1 = wm + 16 + lh;
  const int nB  = wn + lh;

  // A staging: 128 rows x 32 halves = 512 16B-chunks, 2 per thread.
  const int ar0 = tid >> 2, as = (tid & 3) * 8;
  const int ar1 = ar0 + 64;
  int gr0 = m0 + ar0; if (gr0 >= M) gr0 = M - 1;   // clamp: garbage rows unstored
  int gr1 = m0 + ar1; if (gr1 >= M) gr1 = M - 1;
  // B per-step staging (f32 path): 64 cols x 32 halves, 1 chunk per thread.
  const int bc = tid >> 2, bs = (tid & 3) * 8;

  const unsigned ldsA0 = (unsigned)(uintptr_t)&sA[ar0 * 32 + as];
  const unsigned ldsA1 = (unsigned)(uintptr_t)&sA[ar1 * 32 + as];

  const int ksteps = K >> 5;

  if (A16) {
    const _Float16* A = (const _Float16*)Av;
    // ---- prologue: load entire B panel (64 x 256 halves, contiguous both
    // sides: Bt + n0*256 + c*8 <-> sB + c*8) with 8 async b128 copies ----
    {
      const unsigned ldsBp = (unsigned)(uintptr_t)&sB[tid * 8];
      const _Float16* gBp = Bt + (size_t)n0 * 256 + (size_t)tid * 8;
#pragma unroll
      for (int i = 0; i < 8; ++i) {
        asm volatile("global_load_async_to_lds_b128 %0, %1, off"
                     :: "v"(ldsBp + (unsigned)(i * 4096)), "v"(gBp + (size_t)i * 2048)
                     : "memory");
      }
    }
    // ---- prologue: A tile 0 into buffer 0 ----
    {
      const _Float16* ga0 = A + (size_t)gr0 * K + as;
      const _Float16* ga1 = A + (size_t)gr1 * K + as;
      asm volatile("global_load_async_to_lds_b128 %0, %1, off"
                   :: "v"(ldsA0), "v"(ga0) : "memory");
      asm volatile("global_load_async_to_lds_b128 %0, %1, off"
                   :: "v"(ldsA1), "v"(ga1) : "memory");
    }
    for (int kt = 0; kt < ksteps; ++kt) {
      const int k0 = kt << 5;
      const int cur = kt & 1;
      const bool more = (kt + 1) < ksteps;
      if (more) { // prefetch next A tile into the other buffer
        const unsigned boff = (unsigned)((cur ^ 1) * 8192); // 4096 halves
        const _Float16* ga0 = A + (size_t)gr0 * K + k0 + 32 + as;
        const _Float16* ga1 = A + (size_t)gr1 * K + k0 + 32 + as;
        asm volatile("global_load_async_to_lds_b128 %0, %1, off"
                     :: "v"(ldsA0 + boff), "v"(ga0) : "memory");
        asm volatile("global_load_async_to_lds_b128 %0, %1, off"
                     :: "v"(ldsA1 + boff), "v"(ga1) : "memory");
        asm volatile("s_wait_asynccnt 2" ::: "memory");
      } else {
        asm volatile("s_wait_asynccnt 0" ::: "memory");
      }
      __syncthreads();
      const _Float16* sAc = sA + cur * 4096;
      v16h a0, a1, b0, b1;
      { uint4* p = (uint4*)&a0;
        p[0] = *(const uint4*)&sAc[mA0 * 32 + kbA];
        p[1] = *(const uint4*)&sAc[mA0 * 32 + kbA + 16]; }
      { uint4* p = (uint4*)&a1;
        p[0] = *(const uint4*)&sAc[mA1 * 32 + kbA];
        p[1] = *(const uint4*)&sAc[mA1 * 32 + kbA + 16]; }
      { uint4* p = (uint4*)&b0;
        p[0] = *(const uint4*)&sB[nB * 256 + k0 + kbB];
        p[1] = *(const uint4*)&sB[nB * 256 + k0 + kbB + 16]; }
      { uint4* p = (uint4*)&b1;
        p[0] = *(const uint4*)&sB[(nB + 16) * 256 + k0 + kbB];
        p[1] = *(const uint4*)&sB[(nB + 16) * 256 + k0 + kbB + 16]; }
      c00 = __builtin_amdgcn_wmma_f32_16x16x32_f16(false, a0, false, b0, (short)0, c00, false, false);
      c01 = __builtin_amdgcn_wmma_f32_16x16x32_f16(false, a0, false, b1, (short)0, c01, false, false);
      c10 = __builtin_amdgcn_wmma_f32_16x16x32_f16(false, a1, false, b0, (short)0, c10, false, false);
      c11 = __builtin_amdgcn_wmma_f32_16x16x32_f16(false, a1, false, b1, (short)0, c11, false, false);
      __syncthreads();
    }
  } else {
    const float* A = (const float*)Av;
    // prologue: stage tile 0 (A via convert, B via async) into buffer 0
    {
      __align__(16) _Float16 t[8];
      const float* s0 = A + (size_t)gr0 * K + as;
#pragma unroll
      for (int i = 0; i < 8; ++i) t[i] = (_Float16)s0[i];
      *(uint4*)&sA[ar0 * 32 + as] = *(const uint4*)t;
      const float* s1 = A + (size_t)gr1 * K + as;
#pragma unroll
      for (int i = 0; i < 8; ++i) t[i] = (_Float16)s1[i];
      *(uint4*)&sA[ar1 * 32 + as] = *(const uint4*)t;
      const unsigned ldsB = (unsigned)(uintptr_t)&sB[bc * 32 + bs];
      const _Float16* gb = Bt + (size_t)(n0 + bc) * K + bs;
      asm volatile("global_load_async_to_lds_b128 %0, %1, off"
                   :: "v"(ldsB), "v"(gb) : "memory");
    }
    for (int kt = 0; kt < ksteps; ++kt) {
      const int k0 = kt << 5;
      const int cur = kt & 1;
      const bool more = (kt + 1) < ksteps;
      if (more) { // stage next tile into other buffer
        const int nb = cur ^ 1;
        __align__(16) _Float16 t[8];
        const float* s0 = A + (size_t)gr0 * K + k0 + 32 + as;
#pragma unroll
        for (int i = 0; i < 8; ++i) t[i] = (_Float16)s0[i];
        *(uint4*)&sA[nb * 4096 + ar0 * 32 + as] = *(const uint4*)t;
        const float* s1 = A + (size_t)gr1 * K + k0 + 32 + as;
#pragma unroll
        for (int i = 0; i < 8; ++i) t[i] = (_Float16)s1[i];
        *(uint4*)&sA[nb * 4096 + ar1 * 32 + as] = *(const uint4*)t;
        const unsigned ldsB = (unsigned)(uintptr_t)&sB[nb * 2048 + bc * 32 + bs];
        const _Float16* gb = Bt + (size_t)(n0 + bc) * K + k0 + 32 + bs;
        asm volatile("global_load_async_to_lds_b128 %0, %1, off"
                     :: "v"(ldsB), "v"(gb) : "memory");
        asm volatile("s_wait_asynccnt 1" ::: "memory");
      } else {
        asm volatile("s_wait_asynccnt 0" ::: "memory");
      }
      __syncthreads();
      const _Float16* sAc = sA + cur * 4096;
      const _Float16* sBc = sB + cur * 2048;
      v16h a0, a1, b0, b1;
      { uint4* p = (uint4*)&a0;
        p[0] = *(const uint4*)&sAc[mA0 * 32 + kbA];
        p[1] = *(const uint4*)&sAc[mA0 * 32 + kbA + 16]; }
      { uint4* p = (uint4*)&a1;
        p[0] = *(const uint4*)&sAc[mA1 * 32 + kbA];
        p[1] = *(const uint4*)&sAc[mA1 * 32 + kbA + 16]; }
      { uint4* p = (uint4*)&b0;
        p[0] = *(const uint4*)&sBc[nB * 32 + kbB];
        p[1] = *(const uint4*)&sBc[nB * 32 + kbB + 16]; }
      { uint4* p = (uint4*)&b1;
        p[0] = *(const uint4*)&sBc[(nB + 16) * 32 + kbB];
        p[1] = *(const uint4*)&sBc[(nB + 16) * 32 + kbB + 16]; }
      c00 = __builtin_amdgcn_wmma_f32_16x16x32_f16(false, a0, false, b0, (short)0, c00, false, false);
      c01 = __builtin_amdgcn_wmma_f32_16x16x32_f16(false, a0, false, b1, (short)0, c01, false, false);
      c10 = __builtin_amdgcn_wmma_f32_16x16x32_f16(false, a1, false, b0, (short)0, c10, false, false);
      c11 = __builtin_amdgcn_wmma_f32_16x16x32_f16(false, a1, false, b1, (short)0, c11, false, false);
      __syncthreads();
    }
  }

  // C/D layout: VGPR v, lane l -> M = v + (l>=16 ? 8 : 0), N = l&15
  const int cn0 = n0 + wn + lh;
  const int cn1 = cn0 + 16;
  const int rb0 = m0 + wm + ((lane >> 4) << 3);
  const int rb1 = rb0 + 16;
  float b0e = 0.f, b1e = 0.f;
  if (EPI != 3) { b0e = bias[cn0]; b1e = bias[cn1]; }
#pragma unroll
  for (int v = 0; v < 8; ++v) {
    const int r0 = rb0 + v, r1 = rb1 + v;
    if (r0 < M) {
      float* d = C + (size_t)r0 * Nc + cn0;
      float v0 = c00[v] + b0e, v1 = c01[v] + b1e;
      if (EPI == 1) { v0 = fmaxf(v0, 0.f); v1 = fmaxf(v1, 0.f); }
      if (EPI == 2) { d[0] += v0; d[16] += v1; }
      else          { d[0]  = v0; d[16]  = v1; }
    }
    if (r1 < M) {
      float* d = C + (size_t)r1 * Nc + cn0;
      float v0 = c10[v] + b0e, v1 = c11[v] + b1e;
      if (EPI == 1) { v0 = fmaxf(v0, 0.f); v1 = fmaxf(v1, 0.f); }
      if (EPI == 2) { d[0] += v0; d[16] += v1; }
      else          { d[0]  = v0; d[16]  = v1; }
    }
  }
}

// ---------------------------------------------------------------------------
// Weight transpose + f32->f16 convert: Wt[Nc][K] = (f16) W[K][Nc]^T
// ---------------------------------------------------------------------------
__global__ void wtrans_kernel(const float* __restrict__ W, _Float16* __restrict__ Wt,
                              int K, int Nc) {
  size_t idx = (size_t)blockIdx.x * blockDim.x + threadIdx.x;
  if (idx >= (size_t)K * Nc) return;
  int k = (int)(idx / Nc), n = (int)(idx - (size_t)k * Nc);
  Wt[(size_t)n * K + k] = (_Float16)W[idx];
}

__global__ void cvt16_kernel(const float* __restrict__ a, _Float16* __restrict__ o, size_t n) {
  size_t i = (size_t)blockIdx.x * blockDim.x + threadIdx.x;
  if (i < n) o[i] = (_Float16)a[i];
}
__global__ void relu_cvt16_kernel(const float* __restrict__ a, _Float16* __restrict__ o, size_t n) {
  size_t i = (size_t)blockIdx.x * blockDim.x + threadIdx.x;
  if (i < n) o[i] = (_Float16)fmaxf(a[i], 0.f);
}
__global__ void fill_f32_kernel(float* __restrict__ p, float v, size_t n) {
  size_t i = (size_t)blockIdx.x * blockDim.x + threadIdx.x;
  if (i < n) p[i] = v;
}
__global__ void fill_i32_kernel(int* __restrict__ p, int v, size_t n) {
  size_t i = (size_t)blockIdx.x * blockDim.x + threadIdx.x;
  if (i < n) p[i] = v;
}
__global__ void bcast_bias_kernel(float* __restrict__ o, const float* __restrict__ b, size_t n) {
  size_t i = (size_t)blockIdx.x * blockDim.x + threadIdx.x;
  if (i < n) o[i] = b[i & 255];
}

// ---------------------------------------------------------------------------
// LayerNorm(256) + ReLU -> f16, one wave per row, shuffle reductions.
// ---------------------------------------------------------------------------
__global__ __launch_bounds__(256) void ln_relu16_kernel(
    const float* __restrict__ y, const float* __restrict__ g,
    const float* __restrict__ be, _Float16* __restrict__ out, int N) {
  const int row = (int)(((size_t)blockIdx.x * blockDim.x + threadIdx.x) >> 5);
  const int lane = threadIdx.x & 31;
  if (row >= N) return;
  const float* p = y + (size_t)row * 256 + lane * 8;
  float4 v0 = *(const float4*)p;
  float4 v1 = *(const float4*)(p + 4);
  float s = v0.x + v0.y + v0.z + v0.w + v1.x + v1.y + v1.z + v1.w;
  float q = v0.x * v0.x + v0.y * v0.y + v0.z * v0.z + v0.w * v0.w +
            v1.x * v1.x + v1.y * v1.y + v1.z * v1.z + v1.w * v1.w;
#pragma unroll
  for (int off = 1; off < 32; off <<= 1) {
    s += __shfl_xor(s, off, 32);
    q += __shfl_xor(q, off, 32);
  }
  const float mean = s * (1.f / 256.f);
  const float var  = q * (1.f / 256.f) - mean * mean;
  const float rstd = rsqrtf(var + 1e-5f);
  const int c = lane * 8;
  float4 g0 = *(const float4*)(g + c),  g1 = *(const float4*)(g + c + 4);
  float4 e0 = *(const float4*)(be + c), e1 = *(const float4*)(be + c + 4);
  __align__(16) _Float16 t[8];
  t[0] = (_Float16)fmaxf((v0.x - mean) * rstd * g0.x + e0.x, 0.f);
  t[1] = (_Float16)fmaxf((v0.y - mean) * rstd * g0.y + e0.y, 0.f);
  t[2] = (_Float16)fmaxf((v0.z - mean) * rstd * g0.z + e0.z, 0.f);
  t[3] = (_Float16)fmaxf((v0.w - mean) * rstd * g0.w + e0.w, 0.f);
  t[4] = (_Float16)fmaxf((v1.x - mean) * rstd * g1.x + e1.x, 0.f);
  t[5] = (_Float16)fmaxf((v1.y - mean) * rstd * g1.y + e1.y, 0.f);
  t[6] = (_Float16)fmaxf((v1.z - mean) * rstd * g1.z + e1.z, 0.f);
  t[7] = (_Float16)fmaxf((v1.w - mean) * rstd * g1.w + e1.w, 0.f);
  *(uint4*)(out + (size_t)row * 256 + c) = *(const uint4*)t;
}

// ---------------------------------------------------------------------------
// GATv2 edge kernels. Edges 0..E-1 from arrays, E..E+N-1 are self-loops.
// ---------------------------------------------------------------------------
__global__ __launch_bounds__(256) void gat_score_kernel(
    const float* __restrict__ xl, const float* __restrict__ xr,
    const int* __restrict__ src, const int* __restrict__ dst,
    const float* __restrict__ att, float* __restrict__ ex,
    int* __restrict__ mb, int E, int N) {
  const int e = (int)(((size_t)blockIdx.x * blockDim.x + threadIdx.x) >> 5);
  const int lane = threadIdx.x & 31;
  const int ET = E + N;
  if (e >= ET) return;
  int s, d;
  if (e < E) { s = src[e]; d = dst[e]; } else { s = e - E; d = e - E; }
  const int head = lane >> 3, sub = lane & 7;
  const int c = head * 64 + sub * 8;
  const float4* pl = (const float4*)(xl + (size_t)s * 256 + c);
  const float4* pr = (const float4*)(xr + (size_t)d * 256 + c);
  const float4* pa = (const float4*)(att + c);
  float acc = 0.f;
#pragma unroll
  for (int i = 0; i < 2; ++i) {
    float4 a = pl[i], b = pr[i], w = pa[i];
    float v;
    v = a.x + b.x; v = v > 0.f ? v : 0.2f * v; acc += v * w.x;
    v = a.y + b.y; v = v > 0.f ? v : 0.2f * v; acc += v * w.y;
    v = a.z + b.z; v = v > 0.f ? v : 0.2f * v; acc += v * w.z;
    v = a.w + b.w; v = v > 0.f ? v : 0.2f * v; acc += v * w.w;
  }
  acc += __shfl_xor(acc, 1, 32);
  acc += __shfl_xor(acc, 2, 32);
  acc += __shfl_xor(acc, 4, 32);
  if (sub == 0) {
    ex[(size_t)e * 4 + head] = acc;
    int* addr = mb + (size_t)d * 4 + head;  // float atomic-max via int/uint trick
    int iv = __float_as_int(acc);
    if (iv >= 0) atomicMax(addr, iv);
    else atomicMin((unsigned int*)addr, (unsigned int)iv);
  }
}

__global__ void gat_expsum_kernel(
    float* __restrict__ ex, const int* __restrict__ mb,
    float* __restrict__ den, const int* __restrict__ dst, int E, int N) {
  const size_t idx = (size_t)blockIdx.x * blockDim.x + threadIdx.x;
  const size_t n = (size_t)(E + N) * 4;
  if (idx >= n) return;
  const int e = (int)(idx >> 2), head = (int)(idx & 3);
  const int d = (e < E) ? dst[e] : (e - E);
  const float m = __int_as_float(mb[(size_t)d * 4 + head]);
  const float v = __expf(ex[idx] - m);
  ex[idx] = v;
  atomicAdd(den + (size_t)d * 4 + head, v);
}

__global__ __launch_bounds__(256) void gat_scatter_kernel(
    const float* __restrict__ xl, const float* __restrict__ ex,
    const float* __restrict__ den, const int* __restrict__ src,
    const int* __restrict__ dst, float* __restrict__ out, int E, int N) {
  const int e = (int)(((size_t)blockIdx.x * blockDim.x + threadIdx.x) >> 5);
  const int lane = threadIdx.x & 31;
  const int ET = E + N;
  if (e >= ET) return;
  int s, d;
  if (e < E) { s = src[e]; d = dst[e]; } else { s = e - E; d = e - E; }
  const int head = lane >> 3, sub = lane & 7;
  const int c = head * 64 + sub * 8;
  const float alpha = ex[(size_t)e * 4 + head] /
                      (den[(size_t)d * 4 + head] + 1e-16f);
  const float4* ph = (const float4*)(xl + (size_t)s * 256 + c);
  float4 h0 = ph[0], h1 = ph[1];
  float* po = out + (size_t)d * 256 + c;
  atomicAdd(po + 0, h0.x * alpha);
  atomicAdd(po + 1, h0.y * alpha);
  atomicAdd(po + 2, h0.z * alpha);
  atomicAdd(po + 3, h0.w * alpha);
  atomicAdd(po + 4, h1.x * alpha);
  atomicAdd(po + 5, h1.y * alpha);
  atomicAdd(po + 6, h1.z * alpha);
  atomicAdd(po + 7, h1.w * alpha);
}

// ---------------------------------------------------------------------------
extern "C" void kernel_launch(void* const* d_in, const int* in_sizes, int n_in,
                              void* d_out, int out_size, void* d_ws, size_t ws_size,
                              hipStream_t stream) {
  (void)n_in; (void)out_size; (void)ws_size;
  const float* x       = (const float*)d_in[0];
  const int*   ei0     = (const int*)d_in[1];
  const int*   ei1     = (const int*)d_in[2];
  const float* pre_W0  = (const float*)d_in[3];
  const float* pre_b0  = (const float*)d_in[4];
  const float* pre_g0  = (const float*)d_in[5];
  const float* pre_be0 = (const float*)d_in[6];
  const float* pre_W1  = (const float*)d_in[7];
  const float* pre_b1  = (const float*)d_in[8];
  const float* pre_g1  = (const float*)d_in[9];
  const float* pre_be1 = (const float*)d_in[10];
  const float* gat_Wl  = (const float*)d_in[11];
  const float* gat_Wr  = (const float*)d_in[12];
  const float* gat_att = (const float*)d_in[13];
  const float* gat_bias= (const float*)d_in[14];
  const float* proj_W  = (const float*)d_in[15];
  const float* proj_b  = (const float*)d_in[16];
  const float* head_W0 = (const float*)d_in[17];
  const float* head_b0 = (const float*)d_in[18];
  const float* head_W1 = (const float*)d_in[19];
  const float* head_b1 = (const float*)d_in[20];
  float* out = (float*)d_out;

  const int N  = in_sizes[0] / 1024;
  const int E  = in_sizes[1] / 2;
  const int ET = E + N;
  const size_t N256 = (size_t)N * 256;

  // ---- workspace layout ----
  float* ws  = (float*)d_ws;
  float* acc = ws;                 // [N,256] f32 GEMM out / layer accumulator
  float* xl  = acc + N256;         // [N,256] f32 (later aliased as g16)
  float* xr  = xl + N256;          // [N,256] f32 -> reused as gat_out
  float* ex  = xr + N256;          // [ET,4] scores -> exp
  int*   mb  = (int*)(ex + (size_t)ET * 4);          // [N,4] seg-max bits
  float* den = (float*)(mb + (size_t)N * 4);         // [N,4] seg sum
  _Float16* h16 = (_Float16*)(den + (size_t)N * 4);  // [N,256] f16 activations
  _Float16* wt  = h16 + N256;                        // transposed f16 weights
  _Float16* wt_pre0 = wt;                        // [256,1024]
  _Float16* wt_pre1 = wt_pre0 + 256 * 1024;      // [256,256]
  _Float16* wt_Wl   = wt_pre1 + 256 * 256;       // 4 x [256,256]
  _Float16* wt_Wr   = wt_Wl + 4 * 256 * 256;
  _Float16* wt_proj = wt_Wr + 4 * 256 * 256;
  _Float16* wt_h0   = wt_proj + 4 * 256 * 256;
  _Float16* wt_h1   = wt_h0 + 256 * 256;         // [64,256]
  _Float16* g16 = (_Float16*)xl;                 // gat_out f16 (xl dead by then)

  const dim3 blk(256);
  const dim3 gGemm((N + 127) / 128, 4);      // 256-col outputs
  const dim3 gGemmOut((N + 127) / 128, 1);   // 64-col output
  const int lnBlocks   = (N + 7) / 8;
  const int edgeBlocks = (ET + 7) / 8;
  const int ehBlocks   = (int)(((size_t)ET * 4 + 255) / 256);
  const int nhBlocks   = (int)(((size_t)N * 4 + 255) / 256);
  const int elemBlocks = (int)((N256 + 255) / 256);
  const int wb  = (256 * 256 + 255) / 256;
  const int NEG_INF = (int)0xFF800000u;

  // ---- pre-transpose + f16-convert all weights (tiny, L2-resident) ----
  wtrans_kernel<<<(1024 * 256 + 255) / 256, blk, 0, stream>>>(pre_W0, wt_pre0, 1024, 256);
  wtrans_kernel<<<wb, blk, 0, stream>>>(pre_W1, wt_pre1, 256, 256);
  for (int lt = 0; lt < 4; ++lt) {
    wtrans_kernel<<<wb, blk, 0, stream>>>(gat_Wl + (size_t)lt * 65536, wt_Wl + (size_t)lt * 65536, 256, 256);
    wtrans_kernel<<<wb, blk, 0, stream>>>(gat_Wr + (size_t)lt * 65536, wt_Wr + (size_t)lt * 65536, 256, 256);
    wtrans_kernel<<<wb, blk, 0, stream>>>(proj_W + (size_t)lt * 65536, wt_proj + (size_t)lt * 65536, 256, 256);
  }
  wtrans_kernel<<<wb, blk, 0, stream>>>(head_W0, wt_h0, 256, 256);
  wtrans_kernel<<<(256 * 64 + 255) / 256, blk, 0, stream>>>(head_W1, wt_h1, 256, 64);

  // ---- preprocessor MLP ----
  wmma_gemm<0, false><<<gGemm, blk, 0, stream>>>(x, wt_pre0, pre_b0, acc, N, 1024, 256);
  ln_relu16_kernel<<<lnBlocks, blk, 0, stream>>>(acc, pre_g0, pre_be0, h16, N);
  wmma_gemm<0, true><<<gGemm, blk, 0, stream>>>(h16, wt_pre1, pre_b1, acc, N, 256, 256);
  ln_relu16_kernel<<<lnBlocks, blk, 0, stream>>>(acc, pre_g1, pre_be1, h16, N);

  // ---- hetero GAT stack ----
  const int* eis[2] = {ei0, ei1};
  for (int l = 0; l < 2; ++l) {
    for (int t = 0; t < 2; ++t) {
      const int lt = l * 2 + t;
      const float* at = gat_att + (size_t)lt * 256;
      const float* gb = gat_bias + (size_t)lt * 256;
      const float* pb = proj_b + (size_t)lt * 256;
      const int* src = eis[t];
      const int* dst = eis[t] + E;

      wmma_gemm<3, true><<<gGemm, blk, 0, stream>>>(h16, wt_Wl + (size_t)lt * 65536, nullptr, xl, N, 256, 256);
      wmma_gemm<3, true><<<gGemm, blk, 0, stream>>>(h16, wt_Wr + (size_t)lt * 65536, nullptr, xr, N, 256, 256);
      fill_i32_kernel<<<nhBlocks, blk, 0, stream>>>(mb, NEG_INF, (size_t)N * 4);
      fill_f32_kernel<<<nhBlocks, blk, 0, stream>>>(den, 0.f, (size_t)N * 4);
      gat_score_kernel<<<edgeBlocks, blk, 0, stream>>>(xl, xr, src, dst, at, ex, mb, E, N);
      gat_expsum_kernel<<<ehBlocks, blk, 0, stream>>>(ex, mb, den, dst, E, N);
      // xr dead -> becomes gat_out, initialized with conv bias
      bcast_bias_kernel<<<elemBlocks, blk, 0, stream>>>(xr, gb, N256);
      gat_scatter_kernel<<<edgeBlocks, blk, 0, stream>>>(xl, ex, den, src, dst, xr, E, N);
      // xl dead -> g16 = f16(gat_out) for the projection GEMM
      cvt16_kernel<<<elemBlocks, blk, 0, stream>>>(xr, g16, N256);
      if (t == 0)
        wmma_gemm<0, true><<<gGemm, blk, 0, stream>>>(g16, wt_proj + (size_t)lt * 65536, pb, acc, N, 256, 256);
      else
        wmma_gemm<2, true><<<gGemm, blk, 0, stream>>>(g16, wt_proj + (size_t)lt * 65536, pb, acc, N, 256, 256);
    }
    relu_cvt16_kernel<<<elemBlocks, blk, 0, stream>>>(acc, h16, N256);
  }

  // ---- prediction head ----
  wmma_gemm<1, true><<<gGemm, blk, 0, stream>>>(h16, wt_h0, head_b0, acc, N, 256, 256);
  cvt16_kernel<<<elemBlocks, blk, 0, stream>>>(acc, h16, N256);
  wmma_gemm<0, true><<<gGemmOut, blk, 0, stream>>>(h16, wt_h1, head_b1, out, N, 256, 64);
}
